// CrossAttention_66589172957750
// MI455X (gfx1250) — matrix operand order repaired and checked by reading, not compiled, not run on hardware
//
#include <hip/hip_runtime.h>
#include <hip/hip_bf16.h>
#include <math.h>

// ---------------------------------------------------------------------------
// Cross-attention for MI455X (gfx1250, wave32, WMMA 16x16x32 f16 -> f32 acc)
//   B=2, LQ=4096, LKV=1024, QUERY_DIM=512, CROSS_DIM=768, H=8, DH=64, INNER=512
// Pipeline: Q/K/V projections (WMMA GEMM, f16 out, SCALE folded into Q)
//           -> flash attention (WMMA, DPP16 softmax reductions)
//           -> output projection (WMMA GEMM, f32 + bias)
// ---------------------------------------------------------------------------

typedef __attribute__((ext_vector_type(16))) _Float16 v16h;
typedef __attribute__((ext_vector_type(8)))  _Float16 v8h;
typedef __attribute__((ext_vector_type(4)))  _Float16 v4h;
typedef __attribute__((ext_vector_type(2)))  _Float16 v2h;
typedef __attribute__((ext_vector_type(8)))  float    v8f;

#define WMMA_F32_F16(a, b, c) \
  __builtin_amdgcn_wmma_f32_16x16x32_f16(false, (a), false, (b), (short)0, (c), false, false)

// packed f32x2 -> f16x2 convert (v_cvt_pk_f16_f32 path); bit_cast fixes the
// __fp16-vector vs _Float16-vector type mismatch.
__device__ __forceinline__ v2h cvt_pk(float x, float y) {
  return __builtin_bit_cast(v2h, __builtin_amdgcn_cvt_pkrtz(x, y));
}

__device__ __forceinline__ v16h combine8(v8h lo, v8h hi) {
  v16h r;
#pragma unroll
  for (int i = 0; i < 8; ++i) { r[i] = lo[i]; r[i + 8] = hi[i]; }
  return r;
}

// ---- DPP16 cross-lane butterfly (no LDS, pure VALU) -----------------------
// ctrl: 0xB1 = quad_perm(1,0,3,2)  -> xor 1
//       0x4E = quad_perm(2,3,0,1)  -> xor 2
//       0x141 = row_half_mirror    -> pairs the two 4-groups inside each 8
//       0x140 = row_mirror         -> pairs the two 8-groups inside each 16
template <int CTRL>
__device__ __forceinline__ float dpp_mov(float x) {
  return __int_as_float(__builtin_amdgcn_update_dpp(
      0, __float_as_int(x), CTRL, 0xF, 0xF, true));
}
__device__ __forceinline__ float reduce16_max(float v) {
  v = fmaxf(v, dpp_mov<0xB1>(v));
  v = fmaxf(v, dpp_mov<0x4E>(v));
  v = fmaxf(v, dpp_mov<0x141>(v));
  v = fmaxf(v, dpp_mov<0x140>(v));
  return v;
}
__device__ __forceinline__ float reduce16_sum(float v) {
  v += dpp_mov<0xB1>(v);
  v += dpp_mov<0x4E>(v);
  v += dpp_mov<0x141>(v);
  v += dpp_mov<0x140>(v);
  return v;
}

static constexpr int BATCH = 2;
static constexpr int LQ    = 4096;
static constexpr int LKV   = 1024;
static constexpr int QD    = 512;
static constexpr int HEADS = 8;
static constexpr int DH    = 64;
static constexpr int INNER = 512;
static constexpr float SCALE = 0.125f; // 64^-0.5

// ---------------------------------------------------------------------------
// Generic WMMA GEMM:  C[M,N] = oscale * (A[M,K] (f32) * W[K,N] (f32))
// Block: 256 threads (8 waves), block tile 64(M) x 128(N), K step 32.
// Wave (wm,wn) computes 32x32 via 2x2 WMMA tiles.
// MODE 0: out f16, head-split  dst[((b*H+h)*Lseq + l)*64 + d]
// MODE 1: out f16, head-split transposed dst[((b*H+h)*64 + d)*Lseq + l]
// MODE 2: out f32 + bias, plain [M,N]
// LSH: log2(Lseq) (compile-time -> shift/mask instead of div/mod)
// ---------------------------------------------------------------------------
template <int MODE, int LSH>
__global__ __launch_bounds__(256) void gemm_wmma(
    const float* __restrict__ A, const float* __restrict__ W,
    const float* __restrict__ bias, void* __restrict__ outv,
    int M, int K, int N, float oscale) {
  __shared__ _Float16 Alds[64 * 48];    // [row 0..63][k 0..31], stride 48 (pad)
  __shared__ _Float16 Wlds[128 * 48];   // transposed: [n 0..127][k 0..31], stride 48

  const int tid  = threadIdx.x;
  const int lane = tid & 31;
  const int w    = tid >> 5;
  const int wm   = w & 1;               // 0..1  -> M sub-tile
  const int wn   = w >> 1;              // 0..3  -> N sub-tile
  const int ln   = lane & 15;
  const int kh   = lane >> 4;
  const int mbase = blockIdx.x * 64;
  const int nbase = blockIdx.y * 128;

  v8f c[2][2] = {};

  for (int k0 = 0; k0 < K; k0 += 32) {
    // --- stage A tile (64x32) to LDS as f16: packed cvt + b64 LDS stores
#pragma unroll
    for (int it = 0; it < 2; ++it) {
      int q   = tid + it * 256;
      int row = q >> 3;
      int kq  = (q & 7) * 4;
      float4 f = *(const float4*)(A + (size_t)(mbase + row) * K + k0 + kq);
      v2h p0 = cvt_pk(f.x, f.y);
      v2h p1 = cvt_pk(f.z, f.w);
      v4h pk; pk[0] = p0[0]; pk[1] = p0[1]; pk[2] = p1[0]; pk[3] = p1[1];
      *(v4h*)&Alds[row * 48 + kq] = pk;
    }
    // --- stage W tile (32x128) to LDS transposed (packed cvt, scalar stores)
#pragma unroll
    for (int it = 0; it < 4; ++it) {
      int q    = tid + it * 256;
      int krow = q >> 5;
      int nq   = (q & 31) * 4;
      float4 f = *(const float4*)(W + (size_t)(k0 + krow) * N + nbase + nq);
      v2h p0 = cvt_pk(f.x, f.y);
      v2h p1 = cvt_pk(f.z, f.w);
      Wlds[(nq + 0) * 48 + krow] = p0[0];
      Wlds[(nq + 1) * 48 + krow] = p0[1];
      Wlds[(nq + 2) * 48 + krow] = p1[0];
      Wlds[(nq + 3) * 48 + krow] = p1[1];
    }
    // prefetch next A tile into cache while we compute
    if (k0 + 32 < K)
      __builtin_prefetch(A + (size_t)(mbase + (tid >> 3)) * K + k0 + 32, 0, 3);
    __syncthreads();

    // --- fragments
    // A frag (16x32): lane row = ln; elem e -> k = kh*16+e
    v16h af[2];
#pragma unroll
    for (int s = 0; s < 2; ++s) {
      const _Float16* p = &Alds[(wm * 32 + s * 16 + ln) * 48 + kh * 16];
      af[s] = combine8(*(const v8h*)p, *(const v8h*)(p + 8));
    }
    // B frag (32x16): lane col = ln; elem i -> k = kh*16 + i (contiguous in Wlds)
    v16h bf[2];
#pragma unroll
    for (int nt = 0; nt < 2; ++nt) {
      const _Float16* p = &Wlds[(wn * 32 + nt * 16 + ln) * 48 + kh * 16];
      bf[nt] = *(const v16h*)p;
    }
#pragma unroll
    for (int s = 0; s < 2; ++s)
#pragma unroll
      for (int nt = 0; nt < 2; ++nt)
        c[s][nt] = WMMA_F32_F16(af[s], bf[nt], c[s][nt]);
    __syncthreads();
  }

  // --- epilogue: C layout row = e + 8*(lane>>4), col = lane&15
  const int half = lane >> 4;
#pragma unroll
  for (int s = 0; s < 2; ++s) {
#pragma unroll
    for (int nt = 0; nt < 2; ++nt) {
#pragma unroll
      for (int e = 0; e < 8; ++e) {
        int grow = mbase + wm * 32 + s * 16 + e + 8 * half;
        int gcol = nbase + wn * 32 + nt * 16 + ln;
        float val = c[s][nt][e] * oscale;
        if constexpr (MODE == 2) {
          ((float*)outv)[(size_t)grow * N + gcol] = val + bias[gcol];
        } else {
          int bb = grow >> LSH, l = grow & ((1 << LSH) - 1);
          int hh = gcol >> 6, d = gcol & 63;
          size_t bh = (size_t)bb * HEADS + hh;
          if constexpr (MODE == 0)
            ((_Float16*)outv)[(bh << LSH | l) * DH + d] = (_Float16)val;
          else
            ((_Float16*)outv)[((bh * DH + d) << LSH) | l] = (_Float16)val;
        }
      }
    }
  }
}

// ---------------------------------------------------------------------------
// Flash attention:
//   grid.x = LQ/128 (query tile), grid.y = B*H
//   block = 256 threads = 8 waves; wave owns 16 query rows.
//   Q: [B,H,LQ,64] f16 (pre-scaled by 1/sqrt(dh)) ; K: [B,H,LKV,64] f16 ;
//   V: [B,H,64,LKV] f16 (transposed) ; out: [B,LQ,INNER] f32
// ---------------------------------------------------------------------------
__global__ __launch_bounds__(256) void attn_wmma(
    const _Float16* __restrict__ Q, const _Float16* __restrict__ Km,
    const _Float16* __restrict__ Vt, float* __restrict__ O) {
  __shared__ _Float16 Plds[8 * 16 * 72];  // per-wave 16x64 P tile, stride 72

  const int tid  = threadIdx.x;
  const int lane = tid & 31;
  const int w    = tid >> 5;
  const int ln   = lane & 15;
  const int half = lane >> 4;

  const int bh = blockIdx.y;           // b*HEADS + h
  const int b  = bh >> 3;
  const int h  = bh & 7;
  const int qbase = blockIdx.x * 128 + w * 16;

  const _Float16* Qp = Q  + ((size_t)bh * LQ + qbase) * DH;
  const _Float16* Kp = Km + (size_t)bh * LKV * DH;
  const _Float16* Vp = Vt + (size_t)bh * DH * LKV;
  _Float16* Pw = &Plds[w * 16 * 72];

  // Q fragments held for the whole KV sweep (A-matrix layout, K-dim = dh)
  v16h aq[2];
#pragma unroll
  for (int ks = 0; ks < 2; ++ks) {
    const _Float16* p = Qp + (size_t)ln * DH + ks * 32 + half * 16;
    aq[ks] = combine8(*(const v8h*)p, *(const v8h*)(p + 8));
  }

  float mrow[8], lrow[8];
#pragma unroll
  for (int e = 0; e < 8; ++e) { mrow[e] = -INFINITY; lrow[e] = 0.f; }
  v8f o[4] = {};

  for (int kv0 = 0; kv0 < LKV; kv0 += 64) {
    // ---- S = Q * K^T for 16 rows x 64 keys (4 column tiles)
    v8f s[4];
#pragma unroll
    for (int nt = 0; nt < 4; ++nt) {
      const _Float16* kp = Kp + (size_t)(kv0 + nt * 16 + ln) * DH + half * 16;
      v16h bk0 = *(const v16h*)kp;          // d = half*16 + i
      v16h bk1 = *(const v16h*)(kp + 32);   // d = 32 + half*16 + i
      v8f acc = {};
      acc = WMMA_F32_F16(aq[0], bk0, acc);
      acc = WMMA_F32_F16(aq[1], bk1, acc);
      s[nt] = acc;
    }
    // ---- online softmax (row e+8*half; keys across 16 lanes & nt) via DPP16
#pragma unroll
    for (int e = 0; e < 8; ++e) {
      float tm = fmaxf(fmaxf(s[0][e], s[1][e]), fmaxf(s[2][e], s[3][e]));
      tm = reduce16_max(tm);
      float mnew = fmaxf(mrow[e], tm);
      float corr = __expf(mrow[e] - mnew);
      mrow[e] = mnew;
      float sum = 0.f;
#pragma unroll
      for (int nt = 0; nt < 4; ++nt) {
        float p = __expf(s[nt][e] - mnew);
        s[nt][e] = p;
        sum += p;
      }
      sum = reduce16_sum(sum);
      lrow[e] = lrow[e] * corr + sum;
#pragma unroll
      for (int nt = 0; nt < 4; ++nt) o[nt][e] *= corr;
    }
    // ---- P (C layout) -> LDS -> A layout fragments
#pragma unroll
    for (int nt = 0; nt < 4; ++nt)
#pragma unroll
      for (int e = 0; e < 8; ++e)
        Pw[(e + 8 * half) * 72 + nt * 16 + ln] = (_Float16)s[nt][e];

    v16h ap[2];
#pragma unroll
    for (int ks = 0; ks < 2; ++ks) {
      const _Float16* p = &Pw[ln * 72 + ks * 32 + half * 16];
      ap[ks] = combine8(*(const v8h*)p, *(const v8h*)(p + 8));
    }
    // ---- O += P * V   (V transposed: contiguous keys per (d) row)
#pragma unroll
    for (int nt = 0; nt < 4; ++nt) {
#pragma unroll
      for (int ks = 0; ks < 2; ++ks) {
        const _Float16* vp =
            Vp + (size_t)(nt * 16 + ln) * LKV + kv0 + ks * 32 + half * 16;
        v16h bv = *(const v16h*)vp;
        o[nt] = WMMA_F32_F16(ap[ks], bv, o[nt]);
      }
    }
  }

  // ---- normalize and store f32 [B, LQ, INNER]
#pragma unroll
  for (int e = 0; e < 8; ++e) {
    float inv = 1.0f / lrow[e];
    int q = qbase + e + 8 * half;
#pragma unroll
    for (int nt = 0; nt < 4; ++nt) {
      int col = h * DH + nt * 16 + ln;
      O[((size_t)b * LQ + q) * INNER + col] = o[nt][e] * inv;
    }
  }
}

// ---------------------------------------------------------------------------
extern "C" void kernel_launch(void* const* d_in, const int* in_sizes, int n_in,
                              void* d_out, int out_size, void* d_ws, size_t ws_size,
                              hipStream_t stream) {
  const float* hs  = (const float*)d_in[0];  // [2,4096,512]
  const float* enc = (const float*)d_in[1];  // [2,1024,768]
  const float* Wq  = (const float*)d_in[2];  // [512,512]
  const float* Wk  = (const float*)d_in[3];  // [768,512]
  const float* Wv  = (const float*)d_in[4];  // [768,512]
  const float* Wo  = (const float*)d_in[5];  // [512,512]
  const float* bo  = (const float*)d_in[6];  // [512]

  char* ws = (char*)d_ws;
  _Float16* Qf = (_Float16*)(ws);                       // 8 MiB  [B,H,LQ,64]
  _Float16* Kf = (_Float16*)(ws + (8u << 20));          // 2 MiB  [B,H,LKV,64]
  _Float16* Vf = (_Float16*)(ws + (10u << 20));         // 2 MiB  [B,H,64,LKV]
  float*    AO = (float*)   (ws + (12u << 20));         // 16 MiB [B,LQ,512]

  dim3 blk(256);
  // Q = (hs @ Wq) * SCALE      (M=8192, K=512, N=512), Lseq=4096 (LSH=12)
  gemm_wmma<0, 12><<<dim3((BATCH * LQ) / 64, INNER / 128), blk, 0, stream>>>(
      hs, Wq, nullptr, Qf, BATCH * LQ, QD, INNER, SCALE);
  // K = enc @ Wk               (M=2048, K=768, N=512), Lseq=1024 (LSH=10)
  gemm_wmma<0, 10><<<dim3((BATCH * LKV) / 64, INNER / 128), blk, 0, stream>>>(
      enc, Wk, nullptr, Kf, BATCH * LKV, 768, INNER, 1.0f);
  // V = enc @ Wv  (transposed head-split output)
  gemm_wmma<1, 10><<<dim3((BATCH * LKV) / 64, INNER / 128), blk, 0, stream>>>(
      enc, Wv, nullptr, Vf, BATCH * LKV, 768, INNER, 1.0f);
  // attention
  attn_wmma<<<dim3(LQ / 128, BATCH * HEADS), blk, 0, stream>>>(Qf, Kf, Vf, AO);
  // out = AO @ Wo + bo   (f32 output), LSH unused in mode 2
  gemm_wmma<2, 12><<<dim3((BATCH * LQ) / 64, QD / 128), blk, 0, stream>>>(
      AO, Wo, bo, (float*)d_out, BATCH * LQ, INNER, QD, 1.0f);
}
